// CalculateSimilarity_7791070675758
// MI455X (gfx1250) — matrix operand order repaired and checked by reading, not compiled, not run on hardware
//
#include <hip/hip_runtime.h>
#include <hip/hip_bf16.h>
#include <stdint.h>

// ---------------------------------------------------------------------------
// RBF similarity: out[b,n] = exp(-gamma * || X[b,:] - S[n,:] ||^2)
//   via  ||x||^2 + ||s||^2 - 2 * (X @ S^T)
//
// MI455X strategy (compute-bound, AI ~123 FLOP/B):
//   * cross-GEMM on V_WMMA_F32_16X16X32_BF16 with fp32->bf16 hi/lo split
//     (3 WMMA products per K-chunk: hi*hi + hi*lo + lo*hi  ~ fp32 accuracy)
//   * 128x128 C-tile per 256-thread block (8 waves; each wave 64x32 = 4x2
//     16x16 accumulators), K-chunks of 32 staged through padded LDS
//   * hardware v_cvt bf16 converts (not bit tricks) to minimize VALU that
//     competes with WMMA co-execution
//   * row norms precomputed into d_ws by a wave-per-row reduction
// ---------------------------------------------------------------------------

typedef __attribute__((ext_vector_type(16))) __bf16 v16bf;
typedef __attribute__((ext_vector_type(8)))  float  v8f;

#define B_DIM 8192
#define N_DIM 4096
#define D_DIM 256
#define TILE_M 128
#define TILE_N 128
#define TILE_K 32
#define LDS_STRIDE_U 20   // 32 bf16 (16 dwords) + 4 dwords pad -> 80B row pitch

union Frag {
  uint4  q[2];
  v16bf  v;
};

// x ~= hi + lo ; hi = RNE(bf16(x)) via hardware cvt, lo = RNE(bf16(x - hi)).
// (x - (float)hi is exact in fp32, so hi+lo carries ~16 mantissa bits.)
__device__ __forceinline__ void split_bf(float x, unsigned short& hbits,
                                         unsigned short& lbits) {
  __bf16 h = (__bf16)x;          // v_cvt_*bf16_f32 (RNE)
  float hf = (float)h;           // exact widen (shift)
  __bf16 l = (__bf16)(x - hf);   // residual, RNE
  hbits = __builtin_bit_cast(unsigned short, h);
  lbits = __builtin_bit_cast(unsigned short, l);
}

// ---------------------------------------------------------------------------
// Row squared-norm: one wave per row of a (rows x 256) fp32 matrix.
// ---------------------------------------------------------------------------
__global__ __launch_bounds__(256) void row_sqnorm_kernel(
    const float* __restrict__ src, float* __restrict__ dst, int rows) {
  const int wave = threadIdx.x >> 5;
  const int lane = threadIdx.x & 31;
  const int row  = blockIdx.x * 8 + wave;
  if (row >= rows) return;
  const float* p = src + (size_t)row * D_DIM;
  float s = 0.0f;
#pragma unroll
  for (int i = 0; i < D_DIM / 32; ++i) {
    float v = p[lane + 32 * i];
    s += v * v;
  }
#pragma unroll
  for (int off = 16; off > 0; off >>= 1)
    s += __shfl_xor(s, off, 32);
  if (lane == 0) dst[row] = s;
}

// ---------------------------------------------------------------------------
// Main fused kernel.
// ---------------------------------------------------------------------------
__global__ __launch_bounds__(256) void rbf_wmma_kernel(
    const float* __restrict__ X,   // (8192, 256)
    const float* __restrict__ S,   // (4096, 256)
    const float* __restrict__ xs,  // (8192,)  ||x||^2
    const float* __restrict__ ss,  // (4096,)  ||s||^2
    float* __restrict__ out) {     // (8192, 4096)
  __shared__ __align__(16) uint32_t sXhi[TILE_M * LDS_STRIDE_U];
  __shared__ __align__(16) uint32_t sXlo[TILE_M * LDS_STRIDE_U];
  __shared__ __align__(16) uint32_t sShi[TILE_N * LDS_STRIDE_U];
  __shared__ __align__(16) uint32_t sSlo[TILE_N * LDS_STRIDE_U];
  __shared__ float sXS[TILE_M];
  __shared__ float sSS[TILE_N];

  const int t     = threadIdx.x;
  const int lane  = t & 31;
  const int wave  = t >> 5;
  const int wm    = wave >> 2;               // 0..1 : 64-row slab of C-tile
  const int wn    = wave & 3;                // 0..3 : 32-col slab of C-tile
  const int mBase = blockIdx.y * TILE_M;
  const int nBase = blockIdx.x * TILE_N;

  // Stage the norms for this tile (covered by the first in-loop barrier).
  if (t < TILE_M) sXS[t] = xs[mBase + t];
  else            sSS[t - TILE_M] = ss[nBase + (t - TILE_M)];

  v8f acc[4][2];
#pragma unroll
  for (int i = 0; i < 4; ++i)
#pragma unroll
    for (int j = 0; j < 2; ++j)
      acc[i][j] = (v8f){0.f, 0.f, 0.f, 0.f, 0.f, 0.f, 0.f, 0.f};

  // Cooperative tile-load mapping: 256 threads x 4 passes cover 128 rows x
  // 32 cols of fp32; each thread moves one float4 per matrix per pass,
  // 8 consecutive threads cover one row (128B coalesced).
  const int lr = t >> 3;            // 0..31 base row
  const int lc = (t & 7) * 4;       // element column 0,4,...,28

  // Fragment dword offsets within an LDS row (ISA 7.12.2 layouts):
  //  A (16x32 bf16): lane<16 -> K 0..7 & 16..23 ; lane>=16 -> K 8..15 & 24..31
  //  B (32x16 bf16): lane<16 -> K 0..7 & 16..23 ; lane>=16 -> K 8..15 & 24..31
  const int aOff = (lane >> 4) * 4; // dwords
  const int bOff = (lane >> 4) * 4; // dwords

  for (int kc = 0; kc < D_DIM / TILE_K; ++kc) {
    float4 xv[4], sv[4];
#pragma unroll
    for (int p = 0; p < 4; ++p) {
      const int r = lr + 32 * p;
      xv[p] = *(const float4*)(X + (size_t)(mBase + r) * D_DIM + kc * TILE_K + lc);
      sv[p] = *(const float4*)(S + (size_t)(nBase + r) * D_DIM + kc * TILE_K + lc);
    }

    __syncthreads();  // previous chunk fully consumed before overwrite

#pragma unroll
    for (int p = 0; p < 4; ++p) {
      const int r  = lr + 32 * p;
      const int ui = r * LDS_STRIDE_U + (lc >> 1);
      unsigned short hx, lx, hy, ly, hz, lz, hw, lw;
      split_bf(xv[p].x, hx, lx); split_bf(xv[p].y, hy, ly);
      split_bf(xv[p].z, hz, lz); split_bf(xv[p].w, hw, lw);
      *(uint2*)&sXhi[ui] = make_uint2(hx | ((uint32_t)hy << 16),
                                      hz | ((uint32_t)hw << 16));
      *(uint2*)&sXlo[ui] = make_uint2(lx | ((uint32_t)ly << 16),
                                      lz | ((uint32_t)lw << 16));
      split_bf(sv[p].x, hx, lx); split_bf(sv[p].y, hy, ly);
      split_bf(sv[p].z, hz, lz); split_bf(sv[p].w, hw, lw);
      *(uint2*)&sShi[ui] = make_uint2(hx | ((uint32_t)hy << 16),
                                      hz | ((uint32_t)hw << 16));
      *(uint2*)&sSlo[ui] = make_uint2(lx | ((uint32_t)ly << 16),
                                      lz | ((uint32_t)lw << 16));
    }

    __syncthreads();  // tile visible to all waves

    Frag ah[4], al[4], bh[2], bl[2];
#pragma unroll
    for (int i = 0; i < 4; ++i) {
      const int r = wm * 64 + i * 16 + (lane & 15);
      const uint32_t* ph = &sXhi[r * LDS_STRIDE_U];
      const uint32_t* pl = &sXlo[r * LDS_STRIDE_U];
      ah[i].q[0] = *(const uint4*)(ph + aOff);
      ah[i].q[1] = *(const uint4*)(ph + aOff + 8);
      al[i].q[0] = *(const uint4*)(pl + aOff);
      al[i].q[1] = *(const uint4*)(pl + aOff + 8);
    }
#pragma unroll
    for (int j = 0; j < 2; ++j) {
      const int r = wn * 32 + j * 16 + (lane & 15);
      const uint32_t* ph = &sShi[r * LDS_STRIDE_U];
      const uint32_t* pl = &sSlo[r * LDS_STRIDE_U];
      bh[j].q[0] = *(const uint4*)(ph + bOff);
      bh[j].q[1] = *(const uint4*)(ph + bOff + 8);
      bl[j].q[0] = *(const uint4*)(pl + bOff);
      bl[j].q[1] = *(const uint4*)(pl + bOff + 8);
    }

#pragma unroll
    for (int i = 0; i < 4; ++i) {
#pragma unroll
      for (int j = 0; j < 2; ++j) {
        acc[i][j] = __builtin_amdgcn_wmma_f32_16x16x32_bf16(
            false, ah[i].v, false, bh[j].v, (short)0, acc[i][j], false, false);
        acc[i][j] = __builtin_amdgcn_wmma_f32_16x16x32_bf16(
            false, ah[i].v, false, bl[j].v, (short)0, acc[i][j], false, false);
        acc[i][j] = __builtin_amdgcn_wmma_f32_16x16x32_bf16(
            false, al[i].v, false, bh[j].v, (short)0, acc[i][j], false, false);
      }
    }
  }

  // Epilogue: C/D layout (16x16 f32): lanes give N (lane&15), VGPR v gives
  // M = v + 8*(lane>=16).  sq = ||x||^2 + ||s||^2 - 2*cross ; out = exp(-g*sq)
  const float GAMMA = 0.001f;
#pragma unroll
  for (int i = 0; i < 4; ++i) {
#pragma unroll
    for (int j = 0; j < 2; ++j) {
      const int rl0 = wm * 64 + i * 16 + ((lane >> 4) << 3);
      const int cl  = wn * 32 + j * 16 + (lane & 15);
      const float ssv = sSS[cl];
#pragma unroll
      for (int v = 0; v < 8; ++v) {
        const int rl = rl0 + v;
        float sq = sXS[rl] + ssv - 2.0f * acc[i][j][v];
        sq = fmaxf(sq, 0.0f);
        out[(size_t)(mBase + rl) * N_DIM + (nBase + cl)] = __expf(-GAMMA * sq);
      }
    }
  }
}

// ---------------------------------------------------------------------------
extern "C" void kernel_launch(void* const* d_in, const int* in_sizes, int n_in,
                              void* d_out, int out_size, void* d_ws, size_t ws_size,
                              hipStream_t stream) {
  const float* X = (const float*)d_in[0];  // (8192, 256)
  const float* S = (const float*)d_in[1];  // (4096, 256)
  float* out = (float*)d_out;              // (8192, 4096)

  float* xs = (float*)d_ws;        // 8192 floats
  float* ss = xs + B_DIM;          // 4096 floats  (48 KB total in d_ws)

  row_sqnorm_kernel<<<B_DIM / 8, 256, 0, stream>>>(X, xs, B_DIM);
  row_sqnorm_kernel<<<N_DIM / 8, 256, 0, stream>>>(S, ss, N_DIM);

  dim3 grid(N_DIM / TILE_N, B_DIM / TILE_M);  // (32, 64) blocks
  rbf_wmma_kernel<<<grid, 256, 0, stream>>>(X, S, xs, ss, out);
}
